// Model_58248346468798
// MI455X (gfx1250) — matrix operand order repaired and checked by reading, not compiled, not run on hardware
//
#include <hip/hip_runtime.h>
#include <hip/hip_bf16.h>

// ---------------------------------------------------------------------------
// Emergent-communication sender/receiver LSTM game, forward pass.
// B=128, F=2048, V=10000, E=256, H=512, L=20, D=3.
// All GEMMs run on v_wmma_f32_16x16x32_f16 (f16 in, f32 accumulate).
// Steady-state weights are pre-converted to f16 once per call so the hot
// K-loops issue pure b128 f16 loads with no per-iteration v_cvt traffic.
// ---------------------------------------------------------------------------

typedef _Float16 v16h __attribute__((ext_vector_type(16)));
typedef _Float16 h8   __attribute__((ext_vector_type(8)));
typedef float    v8f  __attribute__((ext_vector_type(8)));
typedef float    f4   __attribute__((ext_vector_type(4)));

#define Bv   128
#define Fv   2048
#define Vv   10000
#define Ev   256
#define Hv   512
#define Lv   20
#define LP1v 21

__device__ __forceinline__ v8f wmma_f16(v16h a, v16h b, v8f c) {
    // D = A(16x32) * B(32x16) + C, f32 accumulate
    return __builtin_amdgcn_wmma_f32_16x16x32_f16(false, a, false, b,
                                                  (short)0, c, false, false);
}

// ---- f32-source fragment loaders (convert on the fly; used by the two
// ---- single-shot GEMMs only) ----------------------------------------------
__device__ __forceinline__ v16h a_frag_f32(const float* __restrict__ rowp, int k, int lane) {
    const int koff = (lane >> 4) << 4;                 // 0 or 16
    const f4* p = (const f4*)(rowp + k + koff);
    f4 x0 = p[0], x1 = p[1];
    const f4* q = (const f4*)(rowp + k + koff + 8);
    f4 y0 = q[0], y1 = q[1];
    v16h r;
#pragma unroll
    for (int i = 0; i < 4; ++i) {
        r[i]      = (_Float16)x0[i];
        r[4 + i]  = (_Float16)x1[i];
        r[8 + i]  = (_Float16)y0[i];
        r[12 + i] = (_Float16)y1[i];
    }
    return r;
}

__device__ __forceinline__ v16h b_frag_f32(const float* __restrict__ rowp, int k, int lane) {
    const int koff = (lane >> 4) << 4;                 // 0 or 16
    const f4* p = (const f4*)(rowp + k + koff);
    f4 x0 = p[0], x1 = p[1], x2 = p[2], x3 = p[3];
    v16h r;
#pragma unroll
    for (int i = 0; i < 4; ++i) {
        r[i]      = (_Float16)x0[i];
        r[4 + i]  = (_Float16)x1[i];
        r[8 + i]  = (_Float16)x2[i];
        r[12 + i] = (_Float16)x3[i];
    }
    return r;
}

// ---- pure-f16 fragment loaders (hot path; 2x global_load_b128 each) -------
// A (16x32): lane<16 holds row (lane&15), K=0..15; lane>=16 K=16..31
__device__ __forceinline__ v16h a_frag16(const _Float16* __restrict__ rowp, int k, int lane) {
    const int koff = (lane >> 4) << 4;                 // 0 or 16
    h8 x = *(const h8*)(rowp + k + koff);
    h8 y = *(const h8*)(rowp + k + koff + 8);
    v16h r;
#pragma unroll
    for (int i = 0; i < 8; ++i) { r[i] = x[i]; r[8 + i] = y[i]; }
    return r;
}

// B (32x16): lane<16 holds col (lane&15) with K=0..15; lane>=16 K=16..31
__device__ __forceinline__ v16h b_frag16(const _Float16* __restrict__ rowp, int k, int lane) {
    const int koff = (lane >> 4) << 4;                 // 0 or 16
    h8 x = *(const h8*)(rowp + k + koff);
    h8 y = *(const h8*)(rowp + k + koff + 8);
    v16h r;
#pragma unroll
    for (int i = 0; i < 8; ++i) { r[i] = x[i]; r[8 + i] = y[i]; }
    return r;
}

__device__ __forceinline__ float sigm_(float x) { return 1.0f / (1.0f + expf(-x)); }

__device__ __forceinline__ unsigned int fkey_(float f) {
    unsigned int u = __float_as_uint(f);
    return (u & 0x80000000u) ? ~u : (u | 0x80000000u);   // order-preserving map
}

// ---------------------------------------------------------------------------
__global__ void k_zero(float* __restrict__ p, long long n) {
    long long i = (long long)blockIdx.x * blockDim.x + threadIdx.x;
    long long s = (long long)gridDim.x * blockDim.x;
    for (; i < n; i += s) p[i] = 0.0f;
}

__global__ void k_cvt(const float* __restrict__ s, _Float16* __restrict__ d, long long n) {
    long long i = (long long)blockIdx.x * blockDim.x + threadIdx.x;
    long long st = (long long)gridDim.x * blockDim.x;
    for (; i < n; i += st) d[i] = (_Float16)s[i];
}

__global__ void k_tokinit(int* __restrict__ tok, const int* __restrict__ start,
                          float* __restrict__ m_out) {
    int b = threadIdx.x;            // 128 threads
    int s = *start;
    tok[b] = s;
    m_out[(size_t)b * LP1v * Vv + s] = 1.0f;    // one-hot at step 0
}

// Y[m,n] = sum_k X[m,k] * W[n,k] + bias[n]  (f32 sources; one wave per tile row)
// Optionally mirrors the result into an f16 buffer (for h0 -> h_s16).
template <int NT>
__global__ __launch_bounds__(32) void k_gemm_bias(
    const float* __restrict__ X, int lda,
    const float* __restrict__ W, int K,
    const float* __restrict__ bias,
    float* __restrict__ Y, int ldy,
    _Float16* __restrict__ Yh) {
    const int lane = threadIdx.x;
    const int m0 = blockIdx.y * 16;
    const int n0 = blockIdx.x * (16 * NT);
    const float* arow = X + (size_t)(m0 + (lane & 15)) * lda;
    const float* wrow[NT];
#pragma unroll
    for (int t = 0; t < NT; ++t)
        wrow[t] = W + (size_t)(n0 + t * 16 + (lane & 15)) * K;
    v8f acc[NT];
#pragma unroll
    for (int t = 0; t < NT; ++t)
#pragma unroll
        for (int i = 0; i < 8; ++i) acc[t][i] = 0.0f;

    for (int k = 0; k < K; k += 32) {
        v16h a = a_frag_f32(arow, k, lane);
#pragma unroll
        for (int t = 0; t < NT; ++t)
            acc[t] = wmma_f16(a, b_frag_f32(wrow[t], k, lane), acc[t]);
    }
    const int colr  = lane & 15;
    const int rbase = m0 + ((lane >> 4) << 3);
#pragma unroll
    for (int t = 0; t < NT; ++t) {
        int col = n0 + t * 16 + colr;
        float bv = bias ? bias[col] : 0.0f;
#pragma unroll
        for (int r = 0; r < 8; ++r) {
            float v = acc[t][r] + bv;
            Y[(size_t)(rbase + r) * ldy + col] = v;
            if (Yh) Yh[(size_t)(rbase + r) * ldy + col] = (_Float16)v;
        }
    }
}

// gates[b,j] = emb16[tok[b],:]·Wih16[j,:] + h16[b,:]·Whh16[j,:] + bih[j]+bhh[j]
template <int NT>
__global__ __launch_bounds__(32) void k_gates16(
    const int* __restrict__ tok,
    const _Float16* __restrict__ emb16,   // [V,E]
    const _Float16* __restrict__ h16,     // [B,H]
    const _Float16* __restrict__ Wih16,   // [4H,E]
    const _Float16* __restrict__ Whh16,   // [4H,H]
    const float* __restrict__ bih, const float* __restrict__ bhh,
    float* __restrict__ gates) {          // [B,4H]
    const int lane = threadIdx.x;
    const int m0 = blockIdx.y * 16;
    const int n0 = blockIdx.x * (16 * NT);
    const int mrow = m0 + (lane & 15);
    const _Float16* arow_e = emb16 + (size_t)tok[mrow] * Ev;   // embedding gather
    const _Float16* arow_h = h16 + (size_t)mrow * Hv;
    const _Float16* wri[NT];
    const _Float16* wrh[NT];
#pragma unroll
    for (int t = 0; t < NT; ++t) {
        int n = n0 + t * 16 + (lane & 15);
        wri[t] = Wih16 + (size_t)n * Ev;
        wrh[t] = Whh16 + (size_t)n * Hv;
    }
    v8f acc[NT];
#pragma unroll
    for (int t = 0; t < NT; ++t)
#pragma unroll
        for (int i = 0; i < 8; ++i) acc[t][i] = 0.0f;

    for (int k = 0; k < Ev; k += 32) {
        v16h a = a_frag16(arow_e, k, lane);
#pragma unroll
        for (int t = 0; t < NT; ++t)
            acc[t] = wmma_f16(a, b_frag16(wri[t], k, lane), acc[t]);
    }
    for (int k = 0; k < Hv; k += 32) {
        v16h a = a_frag16(arow_h, k, lane);
#pragma unroll
        for (int t = 0; t < NT; ++t)
            acc[t] = wmma_f16(a, b_frag16(wrh[t], k, lane), acc[t]);
    }
    const int colr  = lane & 15;
    const int rbase = m0 + ((lane >> 4) << 3);
#pragma unroll
    for (int t = 0; t < NT; ++t) {
        int col = n0 + t * 16 + colr;
        float bv = bih[col] + bhh[col];
#pragma unroll
        for (int r = 0; r < 8; ++r)
            gates[(size_t)(rbase + r) * (4 * Hv) + col] = acc[t][r] + bv;
    }
}

// LSTM pointwise (torch gate order i,f,g,o); writes h in f32+f16, resets keys.
__global__ void k_lstm(const float* __restrict__ gates,
                       float* __restrict__ h, _Float16* __restrict__ h16,
                       float* __restrict__ c,
                       unsigned long long* __restrict__ keys) {
    int idx = blockIdx.x * blockDim.x + threadIdx.x;    // B*H
    int b = idx >> 9, u = idx & (Hv - 1);
    const float* g = gates + (size_t)b * 4 * Hv;
    float gi = g[u], gf = g[u + Hv], gg = g[u + 2 * Hv], go = g[u + 3 * Hv];
    float cn = sigm_(gf) * c[idx] + sigm_(gi) * tanhf(gg);
    c[idx] = cn;
    float hv = sigm_(go) * tanhf(cn);
    h[idx] = hv;
    h16[idx] = (_Float16)hv;
    if (keys != nullptr && idx < Bv) keys[idx] = 0ull;
}

// logits = h16 @ lpW16^T + lpb, add Gumbel(0,1) noise, rowwise argmax via
// packed-key half-wave shuffle reduction + global atomicMax_u64.
template <int NT>
__global__ __launch_bounds__(32) void k_logits16(
    const _Float16* __restrict__ h16,
    const _Float16* __restrict__ lpW16,   // [V,H]
    const float* __restrict__ lpb,        // [V]
    const float* __restrict__ gu,         // gumbel_u for this step, [B,V]
    unsigned long long* __restrict__ keys) {
    const int lane = threadIdx.x;
    const int m0 = blockIdx.y * 16;
    const int n0 = blockIdx.x * (16 * NT);
    const _Float16* arow = h16 + (size_t)(m0 + (lane & 15)) * Hv;
    const _Float16* wrow[NT];
#pragma unroll
    for (int t = 0; t < NT; ++t)
        wrow[t] = lpW16 + (size_t)(n0 + t * 16 + (lane & 15)) * Hv;
    v8f acc[NT];
#pragma unroll
    for (int t = 0; t < NT; ++t)
#pragma unroll
        for (int i = 0; i < 8; ++i) acc[t][i] = 0.0f;

    for (int k = 0; k < Hv; k += 32) {
        v16h a = a_frag16(arow, k, lane);
#pragma unroll
        for (int t = 0; t < NT; ++t)
            acc[t] = wmma_f16(a, b_frag16(wrow[t], k, lane), acc[t]);
    }
    const int colr  = lane & 15;
    const int rbase = m0 + ((lane >> 4) << 3);
#pragma unroll
    for (int r = 0; r < 8; ++r) {
        int brow = rbase + r;
        const float* gur = gu + (size_t)brow * Vv;
        unsigned long long best = 0ull;
#pragma unroll
        for (int t = 0; t < NT; ++t) {
            int col = n0 + t * 16 + colr;
            float u = gur[col];
            float gn = -logf(-logf(u + 1e-10f) + 1e-10f);    // Gumbel(0,1)
            float val = acc[t][r] + lpb[col] + gn;           // argmax-equiv to ref
            unsigned long long key =
                ((unsigned long long)fkey_(val) << 32) |
                (unsigned long long)(0xFFFFFFFFu - (unsigned int)col); // ties->min col
            best = best > key ? best : key;
        }
#pragma unroll
        for (int m = 8; m >= 1; m >>= 1) {
            unsigned long long o = __shfl_xor(best, m, 16);  // half-wave reduce
            best = best > o ? best : o;
        }
        if (colr == 0) atomicMax(&keys[brow], best);
    }
}

__global__ void k_tokfin(const unsigned long long* __restrict__ keys,
                         int* __restrict__ tok_next,
                         float* __restrict__ m_out_step) {
    int b = threadIdx.x;            // 128 threads
    unsigned int col = 0xFFFFFFFFu - (unsigned int)(keys[b] & 0xFFFFFFFFull);
    tok_next[b] = (int)col;
    m_out_step[(size_t)b * LP1v * Vv + col] = 1.0f;
}

// hinge loss + accuracy per batch element (one block per b)
__global__ __launch_bounds__(256) void k_loss(
    const float* __restrict__ target, const float* __restrict__ distr,
    const float* __restrict__ rout,
    float* __restrict__ lossb, float* __restrict__ accb) {
    __shared__ float sred[256];
    const int b = blockIdx.x;
    const float* ro = rout + (size_t)b * Fv;
    float s0 = 0.f, s1 = 0.f, s2 = 0.f, s3 = 0.f;
    for (int k = threadIdx.x; k < Fv; k += 256) {
        float rv = ro[k];
        s0 += target[(size_t)b * Fv + k] * rv;
        s1 += distr[((size_t)0 * Bv + b) * Fv + k] * rv;
        s2 += distr[((size_t)1 * Bv + b) * Fv + k] * rv;
        s3 += distr[((size_t)2 * Bv + b) * Fv + k] * rv;
    }
    float part[4] = {s0, s1, s2, s3};
    float tot[4];
    for (int j = 0; j < 4; ++j) {
        sred[threadIdx.x] = part[j];
        __syncthreads();
        for (int st = 128; st > 0; st >>= 1) {
            if (threadIdx.x < st) sred[threadIdx.x] += sred[threadIdx.x + st];
            __syncthreads();
        }
        tot[j] = sred[0];
        __syncthreads();
    }
    if (threadIdx.x == 0) {
        float ts = tot[0], loss = 0.f;
        bool ok = true;
        for (int d = 1; d <= 3; ++d) {
            loss += fmaxf(0.f, 1.f - ts + tot[d]);
            ok = ok && (ts >= tot[d]);          // jnp.argmax ties -> index 0
        }
        lossb[b] = loss;
        accb[b]  = ok ? 1.0f : 0.0f;
    }
}

__global__ void k_reduce(const float* __restrict__ lossb,
                         const float* __restrict__ accb, float* __restrict__ out2) {
    __shared__ float sl[128], sa[128];
    int i = threadIdx.x;
    sl[i] = lossb[i]; sa[i] = accb[i];
    __syncthreads();
    for (int st = 64; st > 0; st >>= 1) {
        if (i < st) { sl[i] += sl[i + st]; sa[i] += sa[i + st]; }
        __syncthreads();
    }
    if (i == 0) { out2[0] = sl[0] / (float)Bv; out2[1] = sa[0] / (float)Bv; }
}

// ---------------------------------------------------------------------------
extern "C" void kernel_launch(void* const* d_in, const int* in_sizes, int n_in,
                              void* d_out, int out_size, void* d_ws, size_t ws_size,
                              hipStream_t stream) {
    const float* target  = (const float*)d_in[0];
    const float* distr   = (const float*)d_in[1];
    const float* gu      = (const float*)d_in[2];
    const float* s_aff_W = (const float*)d_in[3];
    const float* s_aff_b = (const float*)d_in[4];
    const float* s_Wih   = (const float*)d_in[5];
    const float* s_Whh   = (const float*)d_in[6];
    const float* s_bih   = (const float*)d_in[7];
    const float* s_bhh   = (const float*)d_in[8];
    const float* s_emb   = (const float*)d_in[9];
    const float* s_lpW   = (const float*)d_in[10];
    const float* s_lpb   = (const float*)d_in[11];
    const float* r_emb   = (const float*)d_in[12];
    const float* r_Wih   = (const float*)d_in[13];
    const float* r_Whh   = (const float*)d_in[14];
    const float* r_bih   = (const float*)d_in[15];
    const float* r_bhh   = (const float*)d_in[16];
    const float* r_aff_W = (const float*)d_in[17];
    const float* r_aff_b = (const float*)d_in[18];
    const int*   start   = (const int*)d_in[19];

    float* out   = (float*)d_out;
    float* m_out = out + 2;                        // [B, L+1, V]

    // ---- workspace carve (~30.5 MB; f16 region first, all 16B-aligned) ----
    char* w = (char*)d_ws;
    _Float16* s_emb16 = (_Float16*)w;  w += (size_t)Vv * Ev * 2;
    _Float16* s_lpW16 = (_Float16*)w;  w += (size_t)Vv * Hv * 2;
    _Float16* s_Wih16 = (_Float16*)w;  w += (size_t)4 * Hv * Ev * 2;
    _Float16* s_Whh16 = (_Float16*)w;  w += (size_t)4 * Hv * Hv * 2;
    _Float16* r_emb16 = (_Float16*)w;  w += (size_t)Vv * Ev * 2;
    _Float16* r_Wih16 = (_Float16*)w;  w += (size_t)4 * Hv * Ev * 2;
    _Float16* r_Whh16 = (_Float16*)w;  w += (size_t)4 * Hv * Hv * 2;
    _Float16* h_s16   = (_Float16*)w;  w += (size_t)Bv * Hv * 2;
    _Float16* h_r16   = (_Float16*)w;  w += (size_t)Bv * Hv * 2;
    float* h_s   = (float*)w;  w += (size_t)Bv * Hv * 4;   // f32 mirrors (sender h unused after f16)
    float* c_s   = (float*)w;  w += (size_t)Bv * Hv * 4;   // c_s, c_r contiguous (zeroed together)
    float* c_r   = (float*)w;  w += (size_t)Bv * Hv * 4;
    float* h_r   = (float*)w;  w += (size_t)Bv * Hv * 4;   // f32 h_r feeds r_out GEMM
    float* gates = (float*)w;  w += (size_t)Bv * 4 * Hv * 4;
    float* rout  = (float*)w;  w += (size_t)Bv * Fv * 4;
    float* lossb = (float*)w;  w += Bv * 4;
    float* accb  = (float*)w;  w += Bv * 4;
    unsigned long long* keys = (unsigned long long*)w; w += Bv * 8;
    int* tok = (int*)w;        w += (size_t)LP1v * Bv * 4;

    // 0) one-time f32 -> f16 weight conversion (reused across all 41 steps)
    k_cvt<<<1024, 256, 0, stream>>>(s_emb, s_emb16, (long long)Vv * Ev);
    k_cvt<<<1024, 256, 0, stream>>>(s_lpW, s_lpW16, (long long)Vv * Hv);
    k_cvt<<<1024, 256, 0, stream>>>(s_Wih, s_Wih16, (long long)4 * Hv * Ev);
    k_cvt<<<1024, 256, 0, stream>>>(s_Whh, s_Whh16, (long long)4 * Hv * Hv);
    k_cvt<<<1024, 256, 0, stream>>>(r_emb, r_emb16, (long long)Vv * Ev);
    k_cvt<<<1024, 256, 0, stream>>>(r_Wih, r_Wih16, (long long)4 * Hv * Ev);
    k_cvt<<<1024, 256, 0, stream>>>(r_Whh, r_Whh16, (long long)4 * Hv * Hv);

    // 1) zero output canvas + recurrent state
    k_zero<<<2048, 256, 0, stream>>>(out, (long long)out_size);
    k_zero<<<256, 256, 0, stream>>>(c_s, 2LL * Bv * Hv);            // c_s, c_r
    k_zero<<<64, 256, 0, stream>>>((float*)h_r16, (long long)Bv * Hv / 2); // h_r16 = 0
    k_tokinit<<<1, Bv, 0, stream>>>(tok, start, m_out);

    // 2) h0 = target @ s_aff_W^T + b   (M=128, N=512, K=2048); mirror to f16
    k_gemm_bias<4><<<dim3(Hv / 64, Bv / 16), 32, 0, stream>>>(
        target, Fv, s_aff_W, Fv, s_aff_b, h_s, Hv, h_s16);

    // 3) sender: 20 steps
    for (int t = 0; t < Lv; ++t) {
        k_gates16<4><<<dim3((4 * Hv) / 64, Bv / 16), 32, 0, stream>>>(
            tok + t * Bv, s_emb16, h_s16, s_Wih16, s_Whh16, s_bih, s_bhh, gates);
        k_lstm<<<(Bv * Hv) / 256, 256, 0, stream>>>(gates, h_s, h_s16, c_s, keys);
        k_logits16<5><<<dim3(Vv / 80, Bv / 16), 32, 0, stream>>>(
            h_s16, s_lpW16, s_lpb, gu + (size_t)t * Bv * Vv, keys);
        k_tokfin<<<1, Bv, 0, stream>>>(keys, tok + (t + 1) * Bv,
                                       m_out + (size_t)(t + 1) * Vv);
    }

    // 4) receiver: 21 steps over [start_tok, tok_1..tok_20]
    for (int t = 0; t < LP1v; ++t) {
        k_gates16<4><<<dim3((4 * Hv) / 64, Bv / 16), 32, 0, stream>>>(
            tok + t * Bv, r_emb16, h_r16, r_Wih16, r_Whh16, r_bih, r_bhh, gates);
        k_lstm<<<(Bv * Hv) / 256, 256, 0, stream>>>(gates, h_r, h_r16, c_r, nullptr);
    }

    // 5) r_out = h_r @ r_aff_W^T + b   (M=128, N=2048, K=512)
    k_gemm_bias<4><<<dim3(Fv / 64, Bv / 16), 32, 0, stream>>>(
        h_r, Hv, r_aff_W, Hv, r_aff_b, rout, Fv, nullptr);

    // 6) hinge loss + accuracy, deterministic two-stage reduction
    k_loss<<<Bv, 256, 0, stream>>>(target, distr, rout, lossb, accb);
    k_reduce<<<1, Bv, 0, stream>>>(lossb, accb, out);
}
